// ImprovedGRUModel_24404004176439
// MI455X (gfx1250) — matrix operand order repaired and checked by reading, not compiled, not run on hardware
//
#include <hip/hip_runtime.h>
#include <hip/hip_bf16.h>
#include <cstdint>

typedef __attribute__((ext_vector_type(16))) __bf16 v16bf;
typedef __attribute__((ext_vector_type(8)))  float  v8f;

#define B_  64
#define T_  2048
#define D_  64
#define H1_ 256
#define H2_ 128

__device__ __forceinline__ __bf16 tobf(float f) {
    unsigned u = __builtin_bit_cast(unsigned, f);
    unsigned r = (u + 0x7FFFu + ((u >> 16) & 1u)) >> 16;
    unsigned short s = (unsigned short)r;
    return __builtin_bit_cast(__bf16, s);
}

__device__ __forceinline__ v8f wmma_bf16(v16bf a, v16bf b, v8f c) {
    return __builtin_amdgcn_wmma_f32_16x16x32_bf16(false, a, false, b, (short)0, c, false, false);
}

__device__ __forceinline__ float sigmoidf_(float x) {
    return 1.0f / (1.0f + __expf(-x));
}

// ---------------------------------------------------------------------------
// Pack x (B,T,D) fp32 -> A-fragments of x^T: rows M = t*B + b, K = D.
// Fragment layout per ISA (16-bit A 16x32): lane = m + 16*((k>>3)&1),
// elem i: k = ktile*32 + (i<8 ? 8*half+i : 16+8*half+(i-8)).
// One thread per (tile, ktile, lane); writes 16 bf16 (32B).
// ---------------------------------------------------------------------------
__global__ void pack_x_A(const float* __restrict__ x, __bf16* __restrict__ xA) {
    int gid = blockIdx.x * blockDim.x + threadIdx.x;
    const int total = (T_ * 4) * 2 * 32;            // 8192 Mtiles * 2 Ktiles * 32 lanes
    if (gid >= total) return;
    int lane = gid & 31;
    int kt   = (gid >> 5) & 1;
    int tile = gid >> 6;                            // 0..8191
    int t = tile >> 2, btile = tile & 3;
    int m = lane & 15, half = lane >> 4;
    int b = btile * 16 + m;
    const float* xp = x + ((size_t)b * T_ + t) * D_;
    __bf16* o = xA + ((size_t)tile * 2 + kt) * 512 + lane * 16;
#pragma unroll
    for (int i = 0; i < 16; i++) {
        int k = (i < 8) ? (kt * 32 + half * 8 + i)
                        : (kt * 32 + 16 + half * 8 + (i - 8));
        o[i] = tobf(xp[k]);
    }
}

// ---------------------------------------------------------------------------
// Pack weight W [Nrows x K] row-major -> B-fragments (32x16 per tile):
// lane holds column n = nt*16 + lane%16, k = kt*32 + (lane/16)*8 + i + (i&8).
// ---------------------------------------------------------------------------
__global__ void pack_W_B(const float* __restrict__ W, __bf16* __restrict__ out,
                         int K, int NT, int KT) {
    int gid = blockIdx.x * blockDim.x + threadIdx.x;
    int total = NT * KT * 32;
    if (gid >= total) return;
    int lane = gid & 31;
    int rest = gid >> 5;
    int kt = rest % KT;
    int nt = rest / KT;
    int n = nt * 16 + (lane & 15);
    int kbase = kt * 32 + (lane >> 4) * 8;
    __bf16* o = out + ((size_t)nt * KT + kt) * 512 + lane * 16;
#pragma unroll
    for (int i = 0; i < 16; i++)
        o[i] = tobf(W[(size_t)n * K + kbase + i + (i & 8)]);
}

// ---------------------------------------------------------------------------
// Generic bf16 WMMA GEMM: out[M][N] = A @ B^T + bias.  One wave per 16x16 tile.
// A: [Mtiles][Ktiles][32][16] bf16 fragments, B: [Ntiles][Ktiles][32][16].
// ---------------------------------------------------------------------------
__global__ __launch_bounds__(256)
void gemm_bf16(const __bf16* __restrict__ A, const __bf16* __restrict__ Bm,
               const float* __restrict__ bias, float* __restrict__ out,
               int Mtiles, int Ntiles, int Ktiles, int N) {
    int wid  = (blockIdx.x * blockDim.x + threadIdx.x) >> 5;
    int lane = threadIdx.x & 31;
    if (wid >= Mtiles * Ntiles) return;
    int tm = wid / Ntiles, tn = wid % Ntiles;
    const v16bf* Av = (const v16bf*)A;
    const v16bf* Bv = (const v16bf*)Bm;
    v8f acc = {};
    for (int k = 0; k < Ktiles; k++) {
        v16bf a = Av[((size_t)tm * Ktiles + k) * 32 + lane];
        v16bf b = Bv[((size_t)tn * Ktiles + k) * 32 + lane];
        acc = wmma_bf16(a, b, acc);
    }
    int half = lane >> 4, nl = lane & 15;
    int ncol = tn * 16 + nl;
    float bv = bias ? bias[ncol] : 0.0f;
#pragma unroll
    for (int j = 0; j < 8; j++) {
        int m = tm * 16 + j + 8 * half;
        out[(size_t)m * N + ncol] = acc[j] + bv;
    }
}

// ---------------------------------------------------------------------------
// Layer-1 recurrent scan. grid=4 batch groups (16 rows each), block=512 (16
// waves).  Wave w owns hidden cols [16w,16w+16): computes r/z/n gh tiles via
// WMMA (K=256 from LDS h fragments, W_hh1 streamed from L2), updates h (f32
// in registers), repacks h as bf16 A-fragments into LDS for the next step,
// and stores BN1(h) A-fragments to global for the layer-2 input GEMM.
// ---------------------------------------------------------------------------
__global__ __launch_bounds__(512)
void scan1(const __bf16* __restrict__ WB, const float* __restrict__ gx,
           const float* __restrict__ bhh, const float* __restrict__ gam,
           const float* __restrict__ bet, const float* __restrict__ mu,
           const float* __restrict__ var, __bf16* __restrict__ h1A) {
    __shared__ __attribute__((aligned(32))) __bf16 hA[8 * 512];   // 16 rows x 256 K
    int bg = blockIdx.x;
    int w = threadIdx.x >> 5;
    int lane = threadIdx.x & 31;
    int half = lane >> 4, nl = lane & 15;
    int colh = w * 16 + nl;                      // hidden unit this lane produces

    for (int i = threadIdx.x; i < 8 * 512; i += 512) hA[i] = tobf(0.0f);
    __syncthreads();

    float bhr = bhh[colh], bhz = bhh[H1_ + colh], bhn = bhh[2 * H1_ + colh];
    float sc = gam[colh] * rsqrtf(var[colh] + 1e-5f);
    float sh = bet[colh] - mu[colh] * sc;
    float h[8] = {0, 0, 0, 0, 0, 0, 0, 0};

    int ktd = colh >> 5, kk = colh & 31;
    int ii = (kk & 7) + ((kk >> 4) & 1) * 8;
    int lhi = ((kk >> 3) & 1) * 16;
    const v16bf* Wv = (const v16bf*)WB;

    for (int t = 0; t < T_; t++) {
        v8f ar = {}, az = {}, an = {};
#pragma unroll
        for (int k = 0; k < 8; k++) {
            v16bf a  = *(const v16bf*)&hA[k * 512 + lane * 16];
            v16bf b0 = Wv[(((size_t)(0  + w)) * 8 + k) * 32 + lane];
            ar = wmma_bf16(a, b0, ar);
            v16bf b1 = Wv[(((size_t)(16 + w)) * 8 + k) * 32 + lane];
            az = wmma_bf16(a, b1, az);
            v16bf b2 = Wv[(((size_t)(32 + w)) * 8 + k) * 32 + lane];
            an = wmma_bf16(a, b2, an);
        }
        __syncthreads();                         // all hA reads done (WAR)

        const float* gp = gx + ((size_t)t * B_ + bg * 16) * (3 * H1_);
        size_t h1base = ((size_t)(t * 4 + bg) * 8 + ktd) * 512;
#pragma unroll
        for (int j = 0; j < 8; j++) {
            int m = j + 8 * half;
            const float* g = gp + (size_t)m * (3 * H1_);
            float r  = sigmoidf_(g[colh] + ar[j] + bhr);
            float z  = sigmoidf_(g[H1_ + colh] + az[j] + bhz);
            float nn = tanhf(g[2 * H1_ + colh] + r * (an[j] + bhn));
            float hn = (1.0f - z) * nn + z * h[j];
            h[j] = hn;
            int l2 = m + lhi;
            hA[ktd * 512 + l2 * 16 + ii] = tobf(hn);               // next-step A frag
            h1A[h1base + l2 * 16 + ii]   = tobf(hn * sc + sh);     // BN1-folded output
        }
        __syncthreads();                         // hA writes visible (RAW)
    }
}

// ---------------------------------------------------------------------------
// Layer-2 recurrent scan + BN2 + running max over time (fused pooling).
// W_hh2 fragments (96KB bf16) fit in LDS: preloaded once with async LDS copies.
// grid=4 batch groups, block=256 (8 waves; wave w owns hidden cols 16w..).
// ---------------------------------------------------------------------------
__global__ __launch_bounds__(256)
void scan2(const __bf16* __restrict__ WB, const float* __restrict__ gx,
           const float* __restrict__ bhh, const float* __restrict__ gam,
           const float* __restrict__ bet, const float* __restrict__ mu,
           const float* __restrict__ var, float* __restrict__ pooled) {
    __shared__ __attribute__((aligned(32))) __bf16 wlds[24 * 4 * 512];  // 96KB
    __shared__ __attribute__((aligned(32))) __bf16 hA[4 * 512];         // 4KB
    int bg = blockIdx.x;
    int w = threadIdx.x >> 5;
    int lane = threadIdx.x & 31;
    int half = lane >> 4, nl = lane & 15;
    int colh = w * 16 + nl;

    // Async bulk copy of all W_hh2 B-fragments into LDS (CDNA5 async path).
    {
        unsigned ldsbase = (unsigned)(uintptr_t)&wlds[0];
        const char* gsrc = (const char*)WB;
        const unsigned total = 24 * 4 * 512 * 2;
        for (unsigned off = threadIdx.x * 16u; off < total; off += 256u * 16u) {
            unsigned la = ldsbase + off;
            const char* ga = gsrc + off;
            asm volatile("global_load_async_to_lds_b128 %0, %1, off"
                         :: "v"(la), "v"(ga) : "memory");
        }
    }
    for (int i = threadIdx.x; i < 4 * 512; i += 256) hA[i] = tobf(0.0f);
    asm volatile("s_wait_asynccnt 0" ::: "memory");
    __syncthreads();

    float bhr = bhh[colh], bhz = bhh[H2_ + colh], bhn = bhh[2 * H2_ + colh];
    float sc = gam[colh] * rsqrtf(var[colh] + 1e-5f);
    float sh = bet[colh] - mu[colh] * sc;
    float h[8]  = {0, 0, 0, 0, 0, 0, 0, 0};
    float mx[8] = {-3.0e38f, -3.0e38f, -3.0e38f, -3.0e38f,
                   -3.0e38f, -3.0e38f, -3.0e38f, -3.0e38f};

    int ktd = colh >> 5, kk = colh & 31;
    int ii = (kk & 7) + ((kk >> 4) & 1) * 8;
    int lhi = ((kk >> 3) & 1) * 16;
    const v16bf* Wv = (const v16bf*)wlds;

    for (int t = 0; t < T_; t++) {
        v8f ar = {}, az = {}, an = {};
#pragma unroll
        for (int k = 0; k < 4; k++) {
            v16bf a  = *(const v16bf*)&hA[k * 512 + lane * 16];
            v16bf b0 = Wv[((0  + w) * 4 + k) * 32 + lane];
            ar = wmma_bf16(a, b0, ar);
            v16bf b1 = Wv[((8  + w) * 4 + k) * 32 + lane];
            az = wmma_bf16(a, b1, az);
            v16bf b2 = Wv[((16 + w) * 4 + k) * 32 + lane];
            an = wmma_bf16(a, b2, an);
        }
        __syncthreads();

        const float* gp = gx + ((size_t)t * B_ + bg * 16) * (3 * H2_);
#pragma unroll
        for (int j = 0; j < 8; j++) {
            int m = j + 8 * half;
            const float* g = gp + (size_t)m * (3 * H2_);
            float r  = sigmoidf_(g[colh] + ar[j] + bhr);
            float z  = sigmoidf_(g[H2_ + colh] + az[j] + bhz);
            float nn = tanhf(g[2 * H2_ + colh] + r * (an[j] + bhn));
            float hn = (1.0f - z) * nn + z * h[j];
            h[j] = hn;
            mx[j] = fmaxf(mx[j], hn * sc + sh);                    // fused BN2 + maxpool
            hA[ktd * 512 + (m + lhi) * 16 + ii] = tobf(hn);
        }
        __syncthreads();
    }
#pragma unroll
    for (int j = 0; j < 8; j++) {
        int b = bg * 16 + j + 8 * half;
        pooled[(size_t)b * H2_ + colh] = mx[j];
    }
}

// ---------------------------------------------------------------------------
// out[b][o] = fc_b[o] + sum_h tanh(pooled[b][h]) * fc_w[o][h]
// ---------------------------------------------------------------------------
__global__ void final_fc(const float* __restrict__ pooled,
                         const float* __restrict__ fcw,
                         const float* __restrict__ fcb,
                         float* __restrict__ out) {
    int tid = blockIdx.x * blockDim.x + threadIdx.x;
    if (tid >= B_ * 10) return;
    int b = tid / 10, o = tid % 10;
    float s = fcb[o];
    for (int hh = 0; hh < H2_; hh++)
        s += tanhf(pooled[(size_t)b * H2_ + hh]) * fcw[(size_t)o * H2_ + hh];
    out[(size_t)b * 10 + o] = s;
}

extern "C" void kernel_launch(void* const* d_in, const int* in_sizes, int n_in,
                              void* d_out, int out_size, void* d_ws, size_t ws_size,
                              hipStream_t stream) {
    const float* x     = (const float*)d_in[0];
    const float* W_ih1 = (const float*)d_in[1];
    const float* W_hh1 = (const float*)d_in[2];
    const float* b_ih1 = (const float*)d_in[3];
    const float* b_hh1 = (const float*)d_in[4];
    const float* g1    = (const float*)d_in[5];
    const float* be1   = (const float*)d_in[6];
    const float* mu1   = (const float*)d_in[7];
    const float* va1   = (const float*)d_in[8];
    const float* W_ih2 = (const float*)d_in[9];
    const float* W_hh2 = (const float*)d_in[10];
    const float* b_ih2 = (const float*)d_in[11];
    const float* b_hh2 = (const float*)d_in[12];
    const float* g2    = (const float*)d_in[13];
    const float* be2   = (const float*)d_in[14];
    const float* mu2   = (const float*)d_in[15];
    const float* va2   = (const float*)d_in[16];
    const float* fcw   = (const float*)d_in[17];
    const float* fcb   = (const float*)d_in[18];
    (void)in_sizes; (void)n_in; (void)out_size; (void)ws_size;

    char* ws = (char*)d_ws;
    const size_t MT = 8192;                                   // M tiles (T*B/16)
    size_t off = 0;
    auto take = [&](size_t bytes) { size_t c = off; off += (bytes + 255) & ~(size_t)255; return c; };
    __bf16* xA    = (__bf16*)(ws + take(MT * 2 * 512 * 2));   // 16.8 MB
    __bf16* Wih1B = (__bf16*)(ws + take((size_t)48 * 2 * 512 * 2));
    __bf16* Whh1B = (__bf16*)(ws + take((size_t)48 * 8 * 512 * 2));
    __bf16* Wih2B = (__bf16*)(ws + take((size_t)24 * 8 * 512 * 2));
    __bf16* Whh2B = (__bf16*)(ws + take((size_t)24 * 4 * 512 * 2));
    float*  gx1   = (float*)(ws + take((size_t)T_ * B_ * 3 * H1_ * 4));  // 402 MB
    __bf16* h1A   = (__bf16*)(ws + take(MT * 8 * 512 * 2));              // 67 MB
    float*  gx2   = (float*)(ws + take((size_t)T_ * B_ * 3 * H2_ * 4));  // 201 MB
    float*  pooled= (float*)(ws + take((size_t)B_ * H2_ * 4));

    // 1. pack operands into wave32 WMMA fragment layouts (fp32 -> bf16)
    pack_x_A<<<2048, 256, 0, stream>>>(x, xA);
    pack_W_B<<<12, 256, 0, stream>>>(W_ih1, Wih1B, D_,  48, 2);
    pack_W_B<<<48, 256, 0, stream>>>(W_hh1, Whh1B, H1_, 48, 8);
    pack_W_B<<<24, 256, 0, stream>>>(W_ih2, Wih2B, H1_, 24, 8);
    pack_W_B<<<12, 256, 0, stream>>>(W_hh2, Whh2B, H2_, 24, 4);

    // 2. layer-1 input projection: gx1[T*B][768] = x^T @ W_ih1^T + b_ih1
    gemm_bf16<<<(8192 * 48) / 8, 256, 0, stream>>>(xA, Wih1B, b_ih1, gx1, 8192, 48, 2, 768);

    // 3. layer-1 recurrence (batch-parallel, h in LDS, BN1 folded into output)
    scan1<<<4, 512, 0, stream>>>(Whh1B, gx1, b_hh1, g1, be1, mu1, va1, h1A);

    // 4. layer-2 input projection: gx2[T*B][384] = bn1(h1) @ W_ih2^T + b_ih2
    gemm_bf16<<<(8192 * 24) / 8, 256, 0, stream>>>(h1A, Wih2B, b_ih2, gx2, 8192, 24, 8, 384);

    // 5. layer-2 recurrence + BN2 + time max-pool fused
    scan2<<<4, 256, 0, stream>>>(Whh2B, gx2, b_hh2, g2, be2, mu2, va2, pooled);

    // 6. tanh + FC head
    final_fc<<<(B_ * 10 + 255) / 256, 256, 0, stream>>>(pooled, fcw, fcb, (float*)d_out);
}
